// Independence_2173253452538
// MI455X (gfx1250) — matrix operand order repaired and checked by reading, not compiled, not run on hardware
//
#include <hip/hip_runtime.h>
#include <hip/hip_bf16.h>
#include <stdint.h>

#define NHEAD  16
#define DK     64
#define DMODEL 1024
#define SEQ    2048
#define BATCH  2
#define ROWS   (BATCH * SEQ)   // 4096

typedef __attribute__((ext_vector_type(16))) __bf16 v16bf;
typedef __attribute__((ext_vector_type(8)))  __bf16 v8bf;
typedef __attribute__((ext_vector_type(8)))  float  v8f;

__device__ __forceinline__ uint16_t f2bf(float f) {
  union { float f; uint32_t u; } x; x.f = f;
  uint32_t r = x.u + 0x7fffu + ((x.u >> 16) & 1u);   // round-to-nearest-even
  return (uint16_t)(r >> 16);
}

__device__ __forceinline__ v16bf cat16(v8bf lo, v8bf hi) {
  return __builtin_shufflevector(lo, hi, 0,1,2,3,4,5,6,7,8,9,10,11,12,13,14,15);
}

__device__ __forceinline__ v8f wmma_bf16(v16bf a, v16bf b, v8f c) {
  return __builtin_amdgcn_wmma_f32_16x16x32_bf16(false, a, false, b, (short)0, c, false, false);
}

// ---------------- LayerNorm(q) -> bf16 ----------------
__global__ void ln_bf16_kernel(const float* __restrict__ x,
                               const float* __restrict__ gamma,
                               const float* __restrict__ beta,
                               __bf16* __restrict__ out) {
  __shared__ float red[256];
  const int row = blockIdx.x;
  const int tid = threadIdx.x;
  const float* xr = x + (size_t)row * DMODEL;
  float4 v = ((const float4*)xr)[tid];               // 256 threads * 4 = 1024
  red[tid] = v.x + v.y + v.z + v.w;
  __syncthreads();
  for (int off = 128; off > 0; off >>= 1) {
    if (tid < off) red[tid] += red[tid + off];
    __syncthreads();
  }
  const float mu = red[0] * (1.0f / DMODEL);
  __syncthreads();
  const float dx = v.x - mu, dy = v.y - mu, dz = v.z - mu, dw = v.w - mu;
  red[tid] = dx*dx + dy*dy + dz*dz + dw*dw;
  __syncthreads();
  for (int off = 128; off > 0; off >>= 1) {
    if (tid < off) red[tid] += red[tid + off];
    __syncthreads();
  }
  const float rstd = rsqrtf(red[0] * (1.0f / DMODEL) + 1e-6f);
  uint16_t* o = (uint16_t*)(out + (size_t)row * DMODEL);
  const int c = tid * 4;
  o[c + 0] = f2bf(dx * rstd * gamma[c + 0] + beta[c + 0]);
  o[c + 1] = f2bf(dy * rstd * gamma[c + 1] + beta[c + 1]);
  o[c + 2] = f2bf(dz * rstd * gamma[c + 2] + beta[c + 2]);
  o[c + 3] = f2bf(dw * rstd * gamma[c + 3] + beta[c + 3]);
}

// ---------------- f32 -> bf16 elementwise ----------------
__global__ void cvt_bf16_kernel(const float* __restrict__ x,
                                __bf16* __restrict__ out, int n) {
  int i = (blockIdx.x * blockDim.x + threadIdx.x) * 4;
  if (i < n) {
    float4 v = *(const float4*)(x + i);
    uint16_t* o = (uint16_t*)out + i;
    o[0] = f2bf(v.x); o[1] = f2bf(v.y); o[2] = f2bf(v.z); o[3] = f2bf(v.w);
  }
}

// ---------------- transpose W [K x N] f32 -> WT [N x K] bf16 ----------------
__global__ void transpose_bf16_kernel(const float* __restrict__ W,
                                      __bf16* __restrict__ WT) {
  __shared__ float tile[16][17];
  const int k0 = blockIdx.x * 16, n0 = blockIdx.y * 16;
  const int tx = threadIdx.x, ty = threadIdx.y;
  tile[ty][tx] = W[(size_t)(k0 + ty) * DMODEL + (n0 + tx)];
  __syncthreads();
  ((uint16_t*)WT)[(size_t)(n0 + ty) * DMODEL + (k0 + tx)] = f2bf(tile[tx][ty]);
}

// ---------------- projection GEMM: Out[ROWS x DMODEL] = A @ WT^T (bf16 WMMA) --
// A: ROWS x DMODEL row-major bf16.  WT: N x K row-major bf16 (pre-transposed W).
// Each wave: 16 rows x 32 cols (2 tiles) with a one-deep software pipeline.
// Live set ~80 VGPRs -> fits the 128-VGPR budget with no scratch spills while
// keeping next K-chunk's loads in flight under the current WMMAs.
__global__ void __launch_bounds__(256) proj_gemm_kernel(
    const __bf16* __restrict__ A,
    const __bf16* __restrict__ WT,
    __bf16* __restrict__ Out,
    float scale) {
  const int lane = threadIdx.x & 31;
  const int wave = threadIdx.x >> 5;
  const int row0 = blockIdx.x * 128 + wave * 16;
  const int col0 = blockIdx.y * 32;

  const int m    = lane & 15;                 // A row within tile
  const int aoff = (lane < 16) ? 0 : 16;      // A K sub-chunk select
  const int n    = lane & 15;                 // B column within tile
  const int kb   = (lane >> 4) * 16;          // B K base

  v8f acc0 = {}, acc1 = {};
  const __bf16* arow = A  + (size_t)(row0 + m) * DMODEL + aoff;
  const __bf16* brow = WT + (size_t)(col0 + n) * DMODEL + kb;

  // pipeline prologue: stage K-chunk 0
  v8bf  alo = *(const v8bf*)(arow);
  v8bf  ahi = *(const v8bf*)(arow + 8);
  v16bf b0  = *(const v16bf*)(brow);
  v16bf b1  = *(const v16bf*)(brow + 16 * DMODEL);

  for (int k0 = 0; k0 < DMODEL - 32; k0 += 32) {
    // issue next K-chunk's loads before consuming the current one
    const int kn = k0 + 32;
    v8bf  alo_n = *(const v8bf*)(arow + kn);
    v8bf  ahi_n = *(const v8bf*)(arow + kn + 8);
    v16bf b0_n  = *(const v16bf*)(brow + kn);
    v16bf b1_n  = *(const v16bf*)(brow + 16 * DMODEL + kn);
    __builtin_prefetch(arow + kn + 64, 0, 1);     // global_prefetch ahead of A

    v16bf a = cat16(alo, ahi);
    acc0 = wmma_bf16(a, b0, acc0);
    acc1 = wmma_bf16(a, b1, acc1);

    alo = alo_n; ahi = ahi_n;
    b0 = b0_n; b1 = b1_n;
  }
  // epilogue: last K-chunk
  {
    v16bf a = cat16(alo, ahi);
    acc0 = wmma_bf16(a, b0, acc0);
    acc1 = wmma_bf16(a, b1, acc1);
  }

  const int srow = row0 + ((lane >> 4) * 8);  // D rows 0-7 / 8-15
  uint16_t* o = (uint16_t*)Out;
#pragma unroll
  for (int j = 0; j < 8; ++j)
    o[(size_t)(srow + j) * DMODEL + (col0 + n)]      = f2bf(acc0[j] * scale);
#pragma unroll
  for (int j = 0; j < 8; ++j)
    o[(size_t)(srow + j) * DMODEL + (col0 + 16 + n)] = f2bf(acc1[j] * scale);
}

// ------------- fused scores (bf16 WMMA) + row softmax -> attn (f32) ----------
// qh, kh: (B, S, H, 64) bf16.  One block per (16-row strip, b*h).
// 8 waves x 16 col-tiles cover 16 x 2048 scores held in 128 KiB dynamic LDS.
// B fragments for tile i+1 are issued before tile i's WMMAs.
__global__ void __launch_bounds__(256) attn_scores_softmax_kernel(
    const __bf16* __restrict__ qh,
    const __bf16* __restrict__ kh,
    float* __restrict__ attn) {
  extern __shared__ float sc[];               // [16][2048]
  __shared__ float red[256];
  __shared__ float rowmax[16];
  __shared__ float rowsum[16];

  const int tid  = threadIdx.x;
  const int lane = tid & 31;
  const int wave = tid >> 5;
  const int row0 = blockIdx.x * 16;
  const int bh   = blockIdx.y;
  const int b    = bh >> 4;
  const int h    = bh & 15;

  // A fragments (shared across all 16 column tiles of this wave), K = 0..63
  const int m    = lane & 15;
  const int aoff = (lane < 16) ? 0 : 16;
  const __bf16* ap =
      qh + ((size_t)(b * SEQ + row0 + m) * NHEAD + h) * DK + aoff;
  v16bf a0 = cat16(*(const v8bf*)(ap),      *(const v8bf*)(ap + 8));
  v16bf a1 = cat16(*(const v8bf*)(ap + 32), *(const v8bf*)(ap + 40));

  const int n    = lane & 15;
  const int kb   = (lane >> 4) * 16;
  const int lrow = (lane >> 4) * 8;

  // base pointer for this wave's first column tile; tiles step 16 rows of kh
  const __bf16* bp =
      kh + ((size_t)(b * SEQ + wave * 256 + n) * NHEAD + h) * DK + kb;
  const size_t tstep = (size_t)16 * NHEAD * DK;   // 16 kh rows per tile

  v16bf b0 = *(const v16bf*)(bp);
  v16bf b1 = *(const v16bf*)(bp + 32);

#pragma unroll
  for (int i = 0; i < 16; ++i) {
    v16bf b0_n, b1_n;
    if (i < 15) {                                  // compile-time in unrolled body
      b0_n = *(const v16bf*)(bp + (size_t)(i + 1) * tstep);
      b1_n = *(const v16bf*)(bp + (size_t)(i + 1) * tstep + 32);
    }
    v8f acc = {};
    acc = wmma_bf16(a0, b0, acc);
    acc = wmma_bf16(a1, b1, acc);
    const int col0 = (wave * 16 + i) * 16;
#pragma unroll
    for (int j = 0; j < 8; ++j)
      sc[(lrow + j) * SEQ + col0 + n] = acc[j];
    if (i < 15) { b0 = b0_n; b1 = b1_n; }
  }
  __syncthreads();

  // --- softmax over 2048 cols, 16 threads per row ---
  const int r   = tid & 15;
  const int sub = tid >> 4;
  float* rowp = sc + r * SEQ;

  float pmax = -3.4e38f;
  for (int j = 0; j < 128; ++j) pmax = fmaxf(pmax, rowp[sub + j * 16]);
  red[tid] = pmax;
  __syncthreads();
  if (sub == 0) {
    float mx = red[r];
    for (int s2 = 1; s2 < 16; ++s2) mx = fmaxf(mx, red[s2 * 16 + r]);
    rowmax[r] = mx;
  }
  __syncthreads();

  const float mx = rowmax[r];
  float psum = 0.0f;
  for (int j = 0; j < 128; ++j) {
    const int c = sub + j * 16;
    const float e = __expf(rowp[c] - mx);
    rowp[c] = e;
    psum += e;
  }
  red[tid] = psum;
  __syncthreads();
  if (sub == 0) {
    float s = 0.0f;
    for (int s2 = 0; s2 < 16; ++s2) s += red[s2 * 16 + r];
    rowsum[r] = 1.0f / s;
  }
  __syncthreads();

  // coalesced single-touch write of 16 x 2048 attn block
  float* out = attn + ((size_t)bh * SEQ + row0) * SEQ;
  for (int j = 0; j < 128; ++j) {
    const int idx = j * 256 + tid;
    const int rr = idx >> 11;         // / 2048
    const int cc = idx & 2047;
    out[(size_t)rr * SEQ + cc] = sc[rr * SEQ + cc] * rowsum[rr];
  }
}

extern "C" void kernel_launch(void* const* d_in, const int* in_sizes, int n_in,
                              void* d_out, int out_size, void* d_ws, size_t ws_size,
                              hipStream_t stream) {
  const float* q     = (const float*)d_in[0];
  const float* k     = (const float*)d_in[1];
  /* v = d_in[2] unused: attn@v discarded by reference */
  const float* gamma = (const float*)d_in[3];
  const float* beta  = (const float*)d_in[4];
  const float* Wq    = (const float*)d_in[5];
  const float* Wk    = (const float*)d_in[6];
  float* out = (float*)d_out;

  char* ws = (char*)d_ws;
  __bf16* qn  = (__bf16*)(ws);                        // 8 MiB
  __bf16* kbf = (__bf16*)(ws +  8388608);             // 8 MiB
  __bf16* WqT = (__bf16*)(ws + 16777216);             // 2 MiB
  __bf16* WkT = (__bf16*)(ws + 18874368);             // 2 MiB
  __bf16* qhp = (__bf16*)(ws + 20971520);             // 8 MiB
  __bf16* khp = (__bf16*)(ws + 29360128);             // 8 MiB

  // residual = q (first output), then attn
  hipMemcpyAsync(out, q, (size_t)ROWS * DMODEL * sizeof(float),
                 hipMemcpyDeviceToDevice, stream);
  float* attn = out + (size_t)ROWS * DMODEL;

  ln_bf16_kernel<<<ROWS, 256, 0, stream>>>(q, gamma, beta, qn);
  cvt_bf16_kernel<<<(ROWS * DMODEL / 4 + 255) / 256, 256, 0, stream>>>(
      k, kbf, ROWS * DMODEL);
  dim3 tg(DMODEL / 16, DMODEL / 16);
  transpose_bf16_kernel<<<tg, dim3(16, 16), 0, stream>>>(Wq, WqT);
  transpose_bf16_kernel<<<tg, dim3(16, 16), 0, stream>>>(Wk, WkT);

  dim3 pg(ROWS / 128, DMODEL / 32);
  proj_gemm_kernel<<<pg, 256, 0, stream>>>(qn,  WqT, qhp, 0.125f); // fold 1/sqrt(64)
  proj_gemm_kernel<<<pg, 256, 0, stream>>>(kbf, WkT, khp, 1.0f);

  attn_scores_softmax_kernel<<<dim3(SEQ / 16, BATCH * NHEAD), 256,
                               16 * SEQ * sizeof(float), stream>>>(qhp, khp, attn);
}
